// FastParallelSSM_55886114455559
// MI455X (gfx1250) — matrix operand order repaired and checked by reading, not compiled, not run on hardware
//
#include <hip/hip_runtime.h>
#include <hip/hip_bf16.h>

// ---- problem dimensions (compile-time) ----
#define DMODEL   1024
#define DSTATE   16
#define DCONV    4
#define DINNER   2048
#define DTRANK   64
#define BATCH    2
#define SEQL     4096
#define NTOK     (BATCH * SEQL)        // 8192 tokens
#define XPROJ_N  (DTRANK + 2 * DSTATE) // 96

typedef float  v2f   __attribute__((ext_vector_type(2)));
typedef float  v8f   __attribute__((ext_vector_type(8)));
typedef __bf16 v16bf __attribute__((ext_vector_type(16)));

#ifndef __has_builtin
#define __has_builtin(x) 0
#endif
#if defined(__gfx1250__) && __has_builtin(__builtin_amdgcn_tensor_load_to_lds)
#define USE_TDM 1
typedef unsigned int u32x4 __attribute__((ext_vector_type(4)));
typedef int          i32x4 __attribute__((ext_vector_type(4)));
typedef int          i32x8 __attribute__((ext_vector_type(8)));
#else
#define USE_TDM 0
#endif

__device__ __forceinline__ float silu_f(float x) {
    return x / (1.0f + __expf(-x));
}
__device__ __forceinline__ float softplus_f(float x) {
    return (x > 20.0f) ? x : __logf(1.0f + __expf(x));
}
__device__ __forceinline__ unsigned short f2bf_rne(float f) {
    unsigned int u = __float_as_uint(f);
    u += 0x7FFFu + ((u >> 16) & 1u);   // round-to-nearest-even
    return (unsigned short)(u >> 16);
}

// f32 -> bf16 conversion, 8 elements/thread
__global__ void k_cvt_bf16x8(const float* __restrict__ in,
                             unsigned short* __restrict__ out) {
    const size_t i = ((size_t)blockIdx.x * blockDim.x + threadIdx.x) * 8;
    const float4 a = *(const float4*)(in + i);
    const float4 b = *(const float4*)(in + i + 4);
    union { unsigned short h[8]; uint4 q; } r;
    r.h[0] = f2bf_rne(a.x); r.h[1] = f2bf_rne(a.y);
    r.h[2] = f2bf_rne(a.z); r.h[3] = f2bf_rne(a.w);
    r.h[4] = f2bf_rne(b.x); r.h[5] = f2bf_rne(b.y);
    r.h[6] = f2bf_rne(b.z); r.h[7] = f2bf_rne(b.w);
    *(uint4*)(out + i) = r.q;
}

// ---------------------------------------------------------------------------
// BF16 GEMM:  C(f32) = A * W^T,  A: MxK bf16 row-major, W: NxK bf16 row-major.
// One wave computes a 16 x (16*NB) strip; K advances 32 per v_wmma_f32_16x16x32_bf16.
// ---------------------------------------------------------------------------
template<int NB>
__global__ void k_gemm_bf16(const unsigned short* __restrict__ A, int lda,
                            const unsigned short* __restrict__ W, int ldw,
                            float* __restrict__ C, int ldc,
                            int K, int n_groups) {
    const int wave = (int)((blockIdx.x * blockDim.x + threadIdx.x) >> 5);
    const int mt = wave / n_groups;
    const int ng = wave % n_groups;
    const int m0 = mt * 16;
    const int n0 = ng * NB * 16;

    const int lane = threadIdx.x & 31;
    const int h    = lane >> 4;
    const int r    = lane & 15;

    const unsigned short* arow = A + (size_t)(m0 + r) * (size_t)lda;
    const unsigned short* wrow[NB];
#pragma unroll
    for (int nb = 0; nb < NB; ++nb)
        wrow[nb] = W + (size_t)(n0 + nb * 16 + r) * (size_t)ldw + 8 * h;

    v8f acc[NB];
#pragma unroll
    for (int nb = 0; nb < NB; ++nb) acc[nb] = (v8f){0.f,0.f,0.f,0.f,0.f,0.f,0.f,0.f};

    for (int k0 = 0; k0 < K; k0 += 32) {
        union { uint4 q[2]; v16bf v; } af, bf;
        af.q[0] = *(const uint4*)(arow + k0 + 8 * h);        // K [8h, 8h+8)
        af.q[1] = *(const uint4*)(arow + k0 + 16 + 8 * h);   // K [16+8h, +8)
#pragma unroll
        for (int nb = 0; nb < NB; ++nb) {
            bf.q[0] = *(const uint4*)(wrow[nb] + k0);        // W[n][8h .. +8)
            bf.q[1] = *(const uint4*)(wrow[nb] + k0 + 16);   // W[n][16+8h .. +8)
            acc[nb] = __builtin_amdgcn_wmma_f32_16x16x32_bf16(
                false, af.v, false, bf.v, (short)0, acc[nb], false, false);
        }
    }

#pragma unroll
    for (int nb = 0; nb < NB; ++nb) {
        float* crow = C + (size_t)(m0 + h * 8) * (size_t)ldc + n0 + nb * 16 + r;
#pragma unroll
        for (int r2 = 0; r2 < 8; ++r2)
            crow[(size_t)r2 * (size_t)ldc] = acc[nb][r2];
    }
}

// ---------------------------------------------------------------------------
// FP32 WMMA tile (V_WMMA_F32_16X16X4_F32) — dt projection (K=64, exactness).
// ---------------------------------------------------------------------------
__device__ __forceinline__ v8f wmma_f32_tile(const float* __restrict__ A, int lda,
                                             const float* __restrict__ W, int ldw,
                                             int m0, int n0, int K) {
    const int lane = threadIdx.x & 31;
    const int half = lane >> 4;
    const int r    = lane & 15;
    const float* arow = A + (size_t)(m0 + r) * (size_t)lda;
    const float* wrow = W + (size_t)(n0 + r) * (size_t)ldw;
    v8f c = {0.f, 0.f, 0.f, 0.f, 0.f, 0.f, 0.f, 0.f};
#pragma unroll 4
    for (int k0 = 0; k0 < K; k0 += 4) {
        const int kk = k0 + half * 2;
        const float2 af = *(const float2*)(arow + kk);
        const float2 bfv = *(const float2*)(wrow + kk);
        v2f a; a.x = af.x;  a.y = af.y;
        v2f b; b.x = bfv.x; b.y = bfv.y;
        c = __builtin_amdgcn_wmma_f32_16x16x4_f32(false, a, false, b,
                                                  (short)0, c, false, false);
    }
    return c;
}

// dt = softplus(x_dbl[:, :64] @ dt_proj_w^T + dt_proj_b)
__global__ void k_gemm_f32_softplus(const float* __restrict__ A, int lda,
                                    const float* __restrict__ W, int ldw,
                                    const float* __restrict__ bias,
                                    float* __restrict__ C, int ldc,
                                    int K, int n_tiles) {
    const int wave = (int)((blockIdx.x * blockDim.x + threadIdx.x) >> 5);
    const int mt = wave / n_tiles;
    const int nt = wave % n_tiles;
    v8f c = wmma_f32_tile(A, lda, W, ldw, mt * 16, nt * 16, K);
    const int lane = threadIdx.x & 31;
    const int half = lane >> 4;
    const int n    = lane & 15;
    const float bn = bias[nt * 16 + n];
    float* crow = C + (size_t)(mt * 16 + half * 8) * (size_t)ldc + nt * 16 + n;
#pragma unroll
    for (int r2 = 0; r2 < 8; ++r2)
        crow[(size_t)r2 * (size_t)ldc] = softplus_f(c[r2] + bn);
}

// depthwise causal conv (k=4) + bias + SiLU over xp = xz[:, :DINNER]
__global__ void k_conv_silu(const float* __restrict__ xz,
                            const float* __restrict__ cw,
                            const float* __restrict__ cb,
                            float* __restrict__ u) {
    const int idx = (int)(blockIdx.x * blockDim.x + threadIdx.x);
    const int d   = idx & (DINNER - 1);
    const int tok = idx >> 11;
    const int b   = tok >> 12;
    const int l   = tok & (SEQL - 1);
    float acc = cb[d];
#pragma unroll
    for (int j = 0; j < DCONV; ++j) {
        const int lj = l - (DCONV - 1) + j;
        if (lj >= 0)
            acc += xz[((size_t)(b * SEQL + lj)) * (size_t)(2 * DINNER) + d] * cw[d * DCONV + j];
    }
    u[idx] = silu_f(acc);
}

// ---------------------------------------------------------------------------
// selective scan: one lane per channel d, 16-state in VGPRs.
// Per-64-timestep chunk, the shared B/C block (x_dbl rows, cols 64..96 =
// 64x32 f32 tile, row stride 96) is staged into LDS by the Tensor Data Mover
// (tensor_load_to_lds + s_wait_tensorcnt), falling back to cooperative
// vector loads when the TDM builtin is unavailable.
// ---------------------------------------------------------------------------
__global__ void k_scan(const float* __restrict__ u,
                       const float* __restrict__ dt,
                       const float* __restrict__ xdbl,
                       const float* __restrict__ xz,
                       const float* __restrict__ A_log,
                       const float* __restrict__ Dp,
                       float* __restrict__ yz) {
    const int d = (int)(blockIdx.x * blockDim.x + threadIdx.x);
    const int b = blockIdx.y;
    __shared__ float sBC[64][2 * DSTATE];   // [tt][0:16]=B, [tt][16:32]=C

    float Arow[DSTATE];
#pragma unroll
    for (int n = 0; n < DSTATE; ++n)
        Arow[n] = -__expf(A_log[d * DSTATE + n]);
    const float Dd = Dp[d];

    float st[DSTATE];
#pragma unroll
    for (int n = 0; n < DSTATE; ++n) st[n] = 0.0f;

    const size_t tokbase = (size_t)b * SEQL;
    for (int t0 = 0; t0 < SEQL; t0 += 64) {
        __syncthreads();   // previous chunk fully consumed
#if USE_TDM
        if (threadIdx.x < 32) {   // wave 0 issues the tensor DMA (wave-uniform)
            // 2D tile: 64 rows x 32 f32, src row stride 96 f32, dst packed.
            const unsigned long long gaddr =
                (unsigned long long)(const void*)(xdbl + (tokbase + (size_t)t0) * XPROJ_N + DTRANK);
            const unsigned int lds_off = (unsigned int)(unsigned long long)(const void*)&sBC[0][0];
            u32x4 g0;
            g0[0] = 1u;                                        // count=1, user D#
            g0[1] = lds_off;                                   // lds_addr (bytes)
            g0[2] = (unsigned int)(gaddr & 0xFFFFFFFFu);       // global_addr[31:0]
            g0[3] = (unsigned int)((gaddr >> 32) & 0x01FFFFFFu)
                  | 0x80000000u;                               // addr[56:32] | type=2
            i32x8 g1;
            g1[0] = (int)0x00020000u;     // workgroup_mask=0, data_size=2 (4B)
            g1[1] = (int)(32u << 16);     // tensor_dim0 = 32 (bits 79:48 lo)
            g1[2] = (int)(64u << 16);     // tensor_dim0 hi=0 | tensor_dim1 = 64
            g1[3] = (int)(32u << 16);     // tensor_dim1 hi=0 | tile_dim0 = 32
            g1[4] = (int)64u;             // tile_dim1 = 64, tile_dim2 = 0
            g1[5] = (int)96u;             // tensor_dim0_stride = 96 (lo 32)
            g1[6] = 0;                    // stride0 hi | tensor_dim1_stride lo
            g1[7] = 0;
            i32x4 gz = {0, 0, 0, 0};      // groups 2/3 unused (2-D tensor)
#if __clang_major__ >= 23
            i32x8 gz8 = {0, 0, 0, 0, 0, 0, 0, 0};
            __builtin_amdgcn_tensor_load_to_lds(g0, g1, gz, gz, gz8, 0);
#else
            __builtin_amdgcn_tensor_load_to_lds(g0, g1, gz, gz, 0);
#endif
#if __has_builtin(__builtin_amdgcn_s_wait_tensorcnt)
            __builtin_amdgcn_s_wait_tensorcnt(0);
#else
            asm volatile("s_wait_tensorcnt 0x0" ::: "memory");
#endif
        }
#else
        for (int e = threadIdx.x; e < 64 * 2 * DSTATE; e += blockDim.x) {
            const int tt = e >> 5;
            const int n  = e & 31;
            sBC[tt][n] = xdbl[(tokbase + (size_t)(t0 + tt)) * XPROJ_N + DTRANK + n];
        }
#endif
        __syncthreads();   // tile visible to all 8 waves
        for (int tt = 0; tt < 64; ++tt) {
            const size_t tok = tokbase + (size_t)(t0 + tt);
            const float ut  = u [tok * DINNER + d];
            const float dtt = dt[tok * DINNER + d];
            const float zt  = xz[tok * (size_t)(2 * DINNER) + DINNER + d];
            const float dtu = dtt * ut;
            float y = 0.0f;
#pragma unroll
            for (int n = 0; n < DSTATE; ++n) {
                const float dA = __expf(dtt * Arow[n]);
                st[n] = dA * st[n] + dtu * sBC[tt][n];
                y += st[n] * sBC[tt][DSTATE + n];
            }
            y += Dd * ut;
            yz[tok * DINNER + d] = y * silu_f(zt);
        }
    }
}

extern "C" void kernel_launch(void* const* d_in, const int* in_sizes, int n_in,
                              void* d_out, int out_size, void* d_ws, size_t ws_size,
                              hipStream_t stream) {
    const float* x         = (const float*)d_in[0];
    const float* in_proj_w = (const float*)d_in[1]; // (4096, 1024)
    const float* conv_w    = (const float*)d_in[2]; // (2048, 1, 4)
    const float* conv_b    = (const float*)d_in[3]; // (2048,)
    const float* x_proj_w  = (const float*)d_in[4]; // (96, 2048)
    const float* dt_proj_w = (const float*)d_in[5]; // (2048, 64)
    const float* dt_proj_b = (const float*)d_in[6]; // (2048,)
    const float* A_log     = (const float*)d_in[7]; // (2048, 16)
    const float* Dp        = (const float*)d_in[8]; // (2048,)
    const float* out_proj_w= (const float*)d_in[9]; // (1024, 2048)
    float* out = (float*)d_out;                     // (2, 4096, 1024)

    // ---- fp32 intermediates ----
    float* ws   = (float*)d_ws;
    float* xz   = ws;                                     // NTOK x 4096
    float* u    = xz   + (size_t)NTOK * (2 * DINNER);     // NTOK x 2048
    float* xdbl = u    + (size_t)NTOK * DINNER;           // NTOK x 96
    float* dt   = xdbl + (size_t)NTOK * XPROJ_N;          // NTOK x 2048
    float* yz   = dt   + (size_t)NTOK * DINNER;           // NTOK x 2048
    // ---- bf16 shadow buffers ----
    unsigned short* bfbase  = (unsigned short*)(yz + (size_t)NTOK * DINNER);
    unsigned short* xbf     = bfbase;                                 // NTOK x 1024
    unsigned short* w_in_bf = xbf     + (size_t)NTOK * DMODEL;        // 4096 x 1024
    unsigned short* ubf     = w_in_bf + (size_t)(2*DINNER) * DMODEL;  // NTOK x 2048
    unsigned short* w_xp_bf = ubf     + (size_t)NTOK * DINNER;        // 96 x 2048
    unsigned short* yzbf    = w_xp_bf + (size_t)XPROJ_N * DINNER;     // NTOK x 2048
    unsigned short* w_out_bf= yzbf    + (size_t)NTOK * DINNER;        // 1024 x 2048

    // 0) convert x and in_proj_w to bf16
    k_cvt_bf16x8<<<((size_t)NTOK * DMODEL / 8) / 256, 256, 0, stream>>>(x, xbf);
    k_cvt_bf16x8<<<((size_t)(2*DINNER) * DMODEL / 8) / 256, 256, 0, stream>>>(in_proj_w, w_in_bf);

    // 1) xz = x @ in_proj_w^T        M=8192 K=1024 N=4096   (bf16 WMMA, NB=4)
    {
        const int n_groups = (2 * DINNER) / (16 * 4);            // 64
        const int waves = (NTOK / 16) * n_groups;                // 32768
        k_gemm_bf16<4><<<waves / 8, 256, 0, stream>>>(xbf, DMODEL, w_in_bf, DMODEL,
                                                      xz, 2 * DINNER, DMODEL, n_groups);
    }
    // 2) u = silu(depthwise_conv(xp) + conv_b)
    k_conv_silu<<<(NTOK * DINNER) / 256, 256, 0, stream>>>(xz, conv_w, conv_b, u);

    // 2b) convert u and x_proj_w to bf16
    k_cvt_bf16x8<<<((size_t)NTOK * DINNER / 8) / 256, 256, 0, stream>>>(u, ubf);
    k_cvt_bf16x8<<<((size_t)XPROJ_N * DINNER / 8) / 256, 256, 0, stream>>>(x_proj_w, w_xp_bf);

    // 3) x_dbl = u @ x_proj_w^T      M=8192 K=2048 N=96     (bf16 WMMA, NB=2)
    {
        const int n_groups = XPROJ_N / (16 * 2);                 // 3
        const int waves = (NTOK / 16) * n_groups;                // 1536
        k_gemm_bf16<2><<<waves / 8, 256, 0, stream>>>(ubf, DINNER, w_xp_bf, DINNER,
                                                      xdbl, XPROJ_N, DINNER, n_groups);
    }
    // 4) dt = softplus(x_dbl[:, :64] @ dt_proj_w^T + dt_proj_b)  (fp32 WMMA, exact)
    {
        const int waves = (NTOK / 16) * (DINNER / 16);           // 65536
        k_gemm_f32_softplus<<<waves / 8, 256, 0, stream>>>(xdbl, XPROJ_N,
                                                           dt_proj_w, DTRANK,
                                                           dt_proj_b,
                                                           dt, DINNER, DTRANK,
                                                           DINNER / 16);
    }
    // 5) selective scan + gate (TDM-staged B/C tiles)
    k_scan<<<dim3(DINNER / 256, BATCH), 256, 0, stream>>>(u, dt, xdbl, xz,
                                                          A_log, Dp, yz);
    // 5b) convert yz and out_proj_w to bf16
    k_cvt_bf16x8<<<((size_t)NTOK * DINNER / 8) / 256, 256, 0, stream>>>(yz, yzbf);
    k_cvt_bf16x8<<<((size_t)DMODEL * DINNER / 8) / 256, 256, 0, stream>>>(out_proj_w, w_out_bf);

    // 6) out = yz @ out_proj_w^T     M=8192 K=2048 N=1024   (bf16 WMMA, NB=4)
    {
        const int n_groups = DMODEL / (16 * 4);                  // 16
        const int waves = (NTOK / 16) * n_groups;                // 8192
        k_gemm_bf16<4><<<waves / 8, 256, 0, stream>>>(yzbf, DINNER, w_out_bf, DINNER,
                                                      out, DMODEL, DINNER, n_groups);
    }
}
